// Grid2MeshEncoder_11991548690710
// MI455X (gfx1250) — matrix operand order repaired and checked by reading, not compiled, not run on hardware
//
#include <hip/hip_runtime.h>

typedef __attribute__((ext_vector_type(16))) __bf16 v16bf;
typedef __attribute__((ext_vector_type(8)))  float  v8f;

union V16U { v16bf v; uint4 q[2]; };

__device__ __forceinline__ unsigned short f2bf(float f) {
  unsigned int u = __float_as_uint(f);
  u += 0x7fffu + ((u >> 16) & 1u);   // round-to-nearest-even
  return (unsigned short)(u >> 16);
}

// One K-slab of a 16x128 (per-wave) GEMM strip using v_wmma_f32_16x16x32_bf16.
// Xs: [MT][XK] bf16 tile in LDS.  Wt: [512][Kw] bf16 (transposed weights) in global.
template<int XK>
__device__ __forceinline__ void gemm_slab(
    const unsigned short* Xs,
    const unsigned short* __restrict__ Wt,
    int Kw, int ks,
    int lane, int m_sub, int n_quad,
    v8f acc[8]) {
  const int hh  = lane >> 4;   // half-wave select
  const int l15 = lane & 15;
  const int arow = m_sub * 16 + l15;
  for (int k0 = 0; k0 < XK; k0 += 32) {
    // A fragment: lanes 0-15 rows M, K={k0..k0+7, k0+16..k0+23};
    //             lanes 16-31 same rows, K={k0+8..15, k0+24..31}.
    V16U A;
    A.q[0] = *(const uint4*)&Xs[arow * XK + k0 + hh * 8];
    A.q[1] = *(const uint4*)&Xs[arow * XK + k0 + 16 + hh * 8];
#pragma unroll
    for (int t = 0; t < 8; ++t) {
      const int col = n_quad * 128 + t * 16 + l15;
      // B fragment: lanes 0-15 K={k0..k0+7, k0+16..k0+23}, lanes 16-31 K={k0+8..15, k0+24..31} (col per lane).
      const unsigned short* bp = &Wt[(size_t)col * Kw + ks + k0 + hh * 8];
      V16U B;
      B.q[0] = *(const uint4*)bp;
      B.q[1] = *(const uint4*)(bp + 16);
      acc[t] = __builtin_amdgcn_wmma_f32_16x16x32_bf16(
          false, A.v, false, B.v, (short)0, acc[t], false, false);
    }
  }
}

// Fused MLP: out = LN(silu(X@W1+b1)@W2+b2)*g+b (+res).
// X is the concat of up to 3 segments, each optionally index-gathered, zero-padded to K1.
template<int K1>
__global__ __launch_bounds__(256) void mlp_fused(
    int M,
    const float* __restrict__ s0, int w0, int ld0, const int* __restrict__ i0,
    const float* __restrict__ s1, int w1, int ld1, const int* __restrict__ i1,
    const float* __restrict__ s2, int w2, int ld2, const int* __restrict__ i2,
    const unsigned short* __restrict__ W1t, const float* __restrict__ b1,
    const unsigned short* __restrict__ W2t, const float* __restrict__ b2,
    const float* __restrict__ gamma, const float* __restrict__ beta,
    const float* __restrict__ res, float* __restrict__ out) {
  constexpr int MT = 32;
  constexpr int KSLAB = (K1 < 256) ? K1 : 256;
  __shared__ unsigned short Xs[MT * KSLAB];
  __shared__ unsigned short Hs[MT * 512];
  __shared__ float s_sum[MT];
  __shared__ float s_sq[MT];

  const int tid = threadIdx.x;
  const int lane = tid & 31;
  const int wv = tid >> 5;
  const int m_sub = wv >> 2;   // 0..1 : 16-row strip
  const int n_quad = wv & 3;   // 0..3 : 128-col strip
  const int hh = lane >> 4;
  const int l15 = lane & 15;
  const int m0 = blockIdx.x * MT;

  if (tid < MT) { s_sum[tid] = 0.f; s_sq[tid] = 0.f; }

  v8f acc[8];
#pragma unroll
  for (int t = 0; t < 8; ++t)
#pragma unroll
    for (int r = 0; r < 8; ++r) acc[t][r] = 0.f;

  // ---------- layer 1 (K1 streamed through LDS in slabs) ----------
  for (int ks = 0; ks < K1; ks += KSLAB) {
    __syncthreads();
    for (int i = tid; i < MT * KSLAB; i += 256) {
      const int r = i / KSLAB;
      const int c = ks + (i % KSLAB);
      const int grow = m0 + r;
      float v = 0.f;
      if (grow < M) {
        if (c < w0)                v = s0[(size_t)(i0 ? i0[grow] : grow) * ld0 + c];
        else if (c < w0 + w1)      v = s1[(size_t)(i1 ? i1[grow] : grow) * ld1 + (c - w0)];
        else if (c < w0 + w1 + w2) v = s2[(size_t)(i2 ? i2[grow] : grow) * ld2 + (c - w0 - w1)];
      }
      Xs[i] = f2bf(v);
    }
    __syncthreads();
    gemm_slab<KSLAB>(Xs, W1t, K1, ks, lane, m_sub, n_quad, acc);
  }

  // epilogue 1: bias + SiLU -> Hs (bf16)
#pragma unroll
  for (int t = 0; t < 8; ++t) {
    const int col = n_quad * 128 + t * 16 + l15;
    const float bb = b1[col];
#pragma unroll
    for (int r = 0; r < 8; ++r) {
      float h = acc[t][r] + bb;
      h = h / (1.f + __expf(-h));
      const int rl = m_sub * 16 + hh * 8 + r;
      Hs[rl * 512 + col] = f2bf(h);
      acc[t][r] = 0.f;
    }
  }
  __syncthreads();

  // ---------- layer 2 (K = 512, whole H tile in LDS) ----------
  gemm_slab<512>(Hs, W2t, 512, 0, lane, m_sub, n_quad, acc);

  // bias + LayerNorm statistics
  float p[8], q[8];
#pragma unroll
  for (int r = 0; r < 8; ++r) { p[r] = 0.f; q[r] = 0.f; }
#pragma unroll
  for (int t = 0; t < 8; ++t) {
    const int col = n_quad * 128 + t * 16 + l15;
    const float bb = b2[col];
#pragma unroll
    for (int r = 0; r < 8; ++r) {
      const float v = acc[t][r] + bb;
      acc[t][r] = v;
      p[r] += v; q[r] += v * v;
    }
  }
#pragma unroll
  for (int m = 1; m < 16; m <<= 1) {
#pragma unroll
    for (int r = 0; r < 8; ++r) {
      p[r] += __shfl_xor(p[r], m, 32);
      q[r] += __shfl_xor(q[r], m, 32);
    }
  }
  if (l15 == 0) {
#pragma unroll
    for (int r = 0; r < 8; ++r) {
      const int rl = m_sub * 16 + hh * 8 + r;
      atomicAdd(&s_sum[rl], p[r]);
      atomicAdd(&s_sq[rl], q[r]);
    }
  }
  __syncthreads();

  // normalize + affine + optional residual -> fp32 out
#pragma unroll
  for (int t = 0; t < 8; ++t) {
    const int col = n_quad * 128 + t * 16 + l15;
    const float gg = gamma[col];
    const float be = beta[col];
#pragma unroll
    for (int r = 0; r < 8; ++r) {
      const int rl = m_sub * 16 + hh * 8 + r;
      const int grow = m0 + rl;
      if (grow < M) {
        const float mu  = s_sum[rl] * (1.f / 512.f);
        const float var = s_sq[rl] * (1.f / 512.f) - mu * mu;
        const float inv = rsqrtf(var + 1e-5f);
        float y = (acc[t][r] - mu) * inv * gg + be;
        if (res) y += res[(size_t)grow * 512 + col];
        out[(size_t)grow * 512 + col] = y;
      }
    }
  }
}

// Transpose+convert weights: src fp32 [K][512] -> dst bf16 [512][Kpad] (zero K-pad).
__global__ void wprep(const float* __restrict__ src, unsigned short* __restrict__ dst,
                      int K, int Kpad) {
  const size_t total = (size_t)512 * Kpad;
  for (size_t i = (size_t)blockIdx.x * blockDim.x + threadIdx.x; i < total;
       i += (size_t)gridDim.x * blockDim.x) {
    const int c = (int)(i / Kpad);
    const int k = (int)(i % Kpad);
    dst[i] = (k < K) ? f2bf(src[(size_t)k * 512 + c]) : (unsigned short)0;
  }
}

// agg[dst[e]][:] += ef[e][:]
__global__ void scatter_add(const float* __restrict__ ef, const int* __restrict__ dst,
                            float* __restrict__ agg, int E) {
  const size_t total = (size_t)E * 512;
  for (size_t i = (size_t)blockIdx.x * blockDim.x + threadIdx.x; i < total;
       i += (size_t)gridDim.x * blockDim.x) {
    const size_t e = i >> 9;
    const int d = (int)(i & 511);
    unsafeAtomicAdd(&agg[(size_t)dst[e] * 512 + d], ef[i]);
  }
}

extern "C" void kernel_launch(void* const* d_in, const int* in_sizes, int n_in,
                              void* d_out, int out_size, void* d_ws, size_t ws_size,
                              hipStream_t stream) {
  (void)in_sizes; (void)n_in; (void)out_size; (void)ws_size;
  constexpr int NG = 65160, NM = 10242, E = 131072, FIN = 474, FGEO = 38;

  const float* x         = (const float*)d_in[0];
  const float* grid_geo  = (const float*)d_in[1];
  const float* mesh_geo  = (const float*)d_in[2];
  const float* edge_feat = (const float*)d_in[3];
  const float* gW1 = (const float*)d_in[4];  const float* gb1 = (const float*)d_in[5];
  const float* gW2 = (const float*)d_in[6];  const float* gb2 = (const float*)d_in[7];
  const float* gLg = (const float*)d_in[8];  const float* gLb = (const float*)d_in[9];
  const float* mW1 = (const float*)d_in[10]; const float* mb1 = (const float*)d_in[11];
  const float* mW2 = (const float*)d_in[12]; const float* mb2 = (const float*)d_in[13];
  const float* mLg = (const float*)d_in[14]; const float* mLb = (const float*)d_in[15];
  const float* eW1 = (const float*)d_in[16]; const float* eb1 = (const float*)d_in[17];
  const float* eW2 = (const float*)d_in[18]; const float* eb2 = (const float*)d_in[19];
  const float* eLg = (const float*)d_in[20]; const float* eLb = (const float*)d_in[21];
  const float* peW1= (const float*)d_in[22]; const float* peb1= (const float*)d_in[23];
  const float* peW2= (const float*)d_in[24]; const float* peb2= (const float*)d_in[25];
  const float* peLg= (const float*)d_in[26]; const float* peLb= (const float*)d_in[27];
  const float* pnW1= (const float*)d_in[28]; const float* pnb1= (const float*)d_in[29];
  const float* pnW2= (const float*)d_in[30]; const float* pnb2= (const float*)d_in[31];
  const float* pnLg= (const float*)d_in[32]; const float* pnLb= (const float*)d_in[33];
  const float* oW1 = (const float*)d_in[34]; const float* ob1 = (const float*)d_in[35];
  const float* oW2 = (const float*)d_in[36]; const float* ob2 = (const float*)d_in[37];
  const float* oLg = (const float*)d_in[38]; const float* oLb = (const float*)d_in[39];
  const int* edge_src = (const int*)d_in[40];
  const int* edge_dst = (const int*)d_in[41];

  char* ws = (char*)d_ws;
  size_t off = 0;
  auto alloc = [&](size_t bytes) -> char* {
    char* p = ws + off;
    off += (bytes + 255) & ~(size_t)255;
    return p;
  };
  unsigned short* gW1t  = (unsigned short*)alloc((size_t)512 * 512 * 2);
  unsigned short* gW2t  = (unsigned short*)alloc((size_t)512 * 512 * 2);
  unsigned short* mW1t  = (unsigned short*)alloc((size_t)512 * 32 * 2);
  unsigned short* mW2t  = (unsigned short*)alloc((size_t)512 * 512 * 2);
  unsigned short* eW1t  = (unsigned short*)alloc((size_t)512 * 32 * 2);
  unsigned short* eW2t  = (unsigned short*)alloc((size_t)512 * 512 * 2);
  unsigned short* peW1t = (unsigned short*)alloc((size_t)512 * 1536 * 2);
  unsigned short* peW2t = (unsigned short*)alloc((size_t)512 * 512 * 2);
  unsigned short* pnW1t = (unsigned short*)alloc((size_t)512 * 1024 * 2);
  unsigned short* pnW2t = (unsigned short*)alloc((size_t)512 * 512 * 2);
  unsigned short* oW1t  = (unsigned short*)alloc((size_t)512 * 512 * 2);
  unsigned short* oW2t  = (unsigned short*)alloc((size_t)512 * 512 * 2);
  float* gn     = (float*)alloc((size_t)NG * 512 * 4);
  float* mn_emb = (float*)alloc((size_t)NM * 512 * 4);
  float* ef     = (float*)alloc((size_t)E * 512 * 4);
  float* agg    = (float*)alloc((size_t)NM * 512 * 4);

  float* out_grid = (float*)d_out;
  float* out_mesh = (float*)d_out + (size_t)NG * 512;

  auto prep = [&](const float* src, unsigned short* dst, int K, int Kpad) {
    const size_t total = (size_t)512 * Kpad;
    wprep<<<(int)((total + 255) / 256), 256, 0, stream>>>(src, dst, K, Kpad);
  };
  prep(gW1, gW1t, 512, 512);   prep(gW2, gW2t, 512, 512);
  prep(mW1, mW1t, 3, 32);      prep(mW2, mW2t, 512, 512);
  prep(eW1, eW1t, 4, 32);      prep(eW2, eW2t, 512, 512);
  prep(peW1, peW1t, 1536, 1536); prep(peW2, peW2t, 512, 512);
  prep(pnW1, pnW1t, 1024, 1024); prep(pnW2, pnW2t, 512, 512);
  prep(oW1, oW1t, 512, 512);   prep(oW2, oW2t, 512, 512);

  hipMemsetAsync(agg, 0, (size_t)NM * 512 * 4, stream);

  const int MT = 32;
  // grid-node embed: gn = MLP([x, grid_geo])  (474 + 38 = 512)
  mlp_fused<512><<<(NG + MT - 1) / MT, 256, 0, stream>>>(NG,
      x, FIN, FIN, nullptr,
      grid_geo, FGEO, FGEO, nullptr,
      nullptr, 0, 1, nullptr,
      gW1t, gb1, gW2t, gb2, gLg, gLb, nullptr, gn);
  // mesh-node embed: mn = MLP(mesh_geo)  (K padded 3->32)
  mlp_fused<32><<<(NM + MT - 1) / MT, 256, 0, stream>>>(NM,
      mesh_geo, 3, 3, nullptr,
      nullptr, 0, 1, nullptr,
      nullptr, 0, 1, nullptr,
      mW1t, mb1, mW2t, mb2, mLg, mLb, nullptr, mn_emb);
  // edge embed: ef = MLP(edge_feat)  (K padded 4->32)
  mlp_fused<32><<<(E + MT - 1) / MT, 256, 0, stream>>>(E,
      edge_feat, 4, 4, nullptr,
      nullptr, 0, 1, nullptr,
      nullptr, 0, 1, nullptr,
      eW1t, eb1, eW2t, eb2, eLg, eLb, nullptr, ef);
  // edge update: ef += MLP([ef, gn[src], mn[dst]])  (K = 1536, gathered concat)
  mlp_fused<1536><<<(E + MT - 1) / MT, 256, 0, stream>>>(E,
      ef, 512, 512, nullptr,
      gn, 512, 512, edge_src,
      mn_emb, 512, 512, edge_dst,
      peW1t, peb1, peW2t, peb2, peLg, peLb, ef, ef);
  // aggregate edges onto mesh nodes
  scatter_add<<<(int)(((size_t)E * 512 + 255) / 256), 256, 0, stream>>>(ef, edge_dst, agg, E);
  // mesh update: mn_out = mn + MLP([mn, agg])  (K = 1024)
  mlp_fused<1024><<<(NM + MT - 1) / MT, 256, 0, stream>>>(NM,
      mn_emb, 512, 512, nullptr,
      agg, 512, 512, nullptr,
      nullptr, 0, 1, nullptr,
      pnW1t, pnb1, pnW2t, pnb2, pnLg, pnLb, mn_emb, out_mesh);
  // grid output: out = MLP(gn) + gn
  mlp_fused<512><<<(NG + MT - 1) / MT, 256, 0, stream>>>(NG,
      gn, 512, 512, nullptr,
      nullptr, 0, 1, nullptr,
      nullptr, 0, 1, nullptr,
      oW1t, ob1, oW2t, ob2, oLg, oLb, gn, out_grid);
}
